// Dynamic_conv2d_16612933500956
// MI455X (gfx1250) — compile-verified, matrix operand order checked by reading, not run on hardware
//
#include <hip/hip_runtime.h>
#include <hip/hip_bf16.h>

typedef unsigned short u16;
typedef unsigned int   u32;

#define B_SZ   32
#define CIN    256
#define COUT   256
#define HW     56
#define NPIX   (HW * HW)        // 3136 = 49 * 64
#define KBANK  4
#define HIDDEN 129
#define TEMP   34.0f
#define LDSTR  40               // LDS row stride in u16 (80 B: 16B-aligned, low-conflict)

typedef __attribute__((ext_vector_type(16))) __bf16 v16bf;
typedef __attribute__((ext_vector_type(8)))  float  v8f;

union BF16x16 { v16bf v; uint4 q[2]; };
union F32x8   { v8f  v; float f[8]; };

__device__ __forceinline__ u16 f32_to_bf16(float f) {
    union { float f; u32 u; } cv; cv.f = f;
    u32 u = cv.u;
    u32 r = u + 0x7FFFu + ((u >> 16) & 1u);   // round-to-nearest-even
    return (u16)(r >> 16);
}

#if __has_builtin(__builtin_amdgcn_cvt_pk_bf16_f32)
typedef __attribute__((ext_vector_type(2))) __bf16 v2bf;
__device__ __forceinline__ u32 pack_bf16(float lo, float hi) {
    union { v2bf v; u32 u; } r;
    r.v = __builtin_amdgcn_cvt_pk_bf16_f32(lo, hi);   // v_cvt_pk_bf16_f32
    return r.u;
}
#else
__device__ __forceinline__ u32 pack_bf16(float lo, float hi) {
    return (u32)f32_to_bf16(lo) | ((u32)f32_to_bf16(hi) << 16);
}
#endif

// ---------------------------------------------------------------------------
// 1) mix pooling: p[b, c] = avg, p[b, 256+c] = max   (one block per (b,c))
// ---------------------------------------------------------------------------
__global__ __launch_bounds__(256) void pool_kernel(const float* __restrict__ x,
                                                   float* __restrict__ p) {
    const int bc = blockIdx.x;            // b*256 + c
    const int b = bc >> 8, c = bc & 255;
    const float* src = x + (size_t)bc * NPIX;
    float s = 0.f, m = -3.4e38f;
    for (int i = threadIdx.x; i < NPIX; i += 256) {
        float v = src[i];
        s += v;
        m = fmaxf(m, v);
    }
    __shared__ float ss[256], sm[256];
    ss[threadIdx.x] = s; sm[threadIdx.x] = m;
    __syncthreads();
    for (int st = 128; st > 0; st >>= 1) {
        if (threadIdx.x < st) {
            ss[threadIdx.x] += ss[threadIdx.x + st];
            sm[threadIdx.x] = fmaxf(sm[threadIdx.x], sm[threadIdx.x + st]);
        }
        __syncthreads();
    }
    if (threadIdx.x == 0) {
        p[b * (2 * CIN) + c]        = ss[0] * (1.f / (float)NPIX);
        p[b * (2 * CIN) + CIN + c]  = sm[0];
    }
}

// ---------------------------------------------------------------------------
// 2) attention MLP + softmax + aggregated bias (single tiny block)
// ---------------------------------------------------------------------------
__global__ __launch_bounds__(256) void attn_kernel(const float* __restrict__ p,
                                                   const float* __restrict__ w1,
                                                   const float* __restrict__ w2,
                                                   const float* __restrict__ b2,
                                                   const float* __restrict__ bias_k,
                                                   float* __restrict__ attn,
                                                   float* __restrict__ aggb) {
    __shared__ float h[B_SZ * HIDDEN];
    __shared__ float logit[B_SZ * KBANK];
    __shared__ float sattn[B_SZ * KBANK];

    for (int idx = threadIdx.x; idx < B_SZ * HIDDEN; idx += 256) {
        const int b = idx / HIDDEN, hh = idx % HIDDEN;
        const float* pb = p  + b  * (2 * CIN);
        const float* wr = w1 + hh * (2 * CIN);
        float acc = 0.f;
        for (int i = 0; i < 2 * CIN; ++i) acc += pb[i] * wr[i];
        h[idx] = fmaxf(acc, 0.f);
    }
    __syncthreads();
    for (int idx = threadIdx.x; idx < B_SZ * KBANK; idx += 256) {
        const int b = idx >> 2, k = idx & 3;
        const float* hb = h  + b * HIDDEN;
        const float* wr = w2 + k * HIDDEN;
        float acc = b2[k];
        for (int i = 0; i < HIDDEN; ++i) acc += hb[i] * wr[i];
        logit[idx] = acc * (1.0f / TEMP);
    }
    __syncthreads();
    if (threadIdx.x < B_SZ) {
        const int b = threadIdx.x;
        float mx = logit[b * 4];
        for (int k = 1; k < 4; ++k) mx = fmaxf(mx, logit[b * 4 + k]);
        float e[4], s = 0.f;
        for (int k = 0; k < 4; ++k) { e[k] = expf(logit[b * 4 + k] - mx); s += e[k]; }
        const float inv = 1.f / s;
        for (int k = 0; k < 4; ++k) {
            sattn[b * 4 + k] = e[k] * inv;
            attn[b * 4 + k]  = e[k] * inv;
        }
    }
    __syncthreads();
    for (int idx = threadIdx.x; idx < B_SZ * COUT; idx += 256) {
        const int b = idx >> 8, o = idx & 255;
        float acc = 0.f;
        for (int k = 0; k < 4; ++k) acc += sattn[b * 4 + k] * bias_k[k * COUT + o];
        aggb[idx] = acc;
    }
}

// ---------------------------------------------------------------------------
// 3) aggregate kernel bank into per-sample bf16 weights, layout [b][o][tap][c]
// ---------------------------------------------------------------------------
__global__ __launch_bounds__(256) void aggw_kernel(const float* __restrict__ weight,
                                                   const float* __restrict__ attn,
                                                   u16* __restrict__ aggw) {
    const long long idx = (long long)blockIdx.x * 256 + threadIdx.x; // [0, 32*256*9*256)
    const int c   = (int)(idx & 255);
    long long r   = idx >> 8;
    const int tap = (int)(r % 9); r /= 9;
    const int o   = (int)(r & 255);
    const int b   = (int)(r >> 8);

    const float a0 = attn[b * 4 + 0];
    const float a1 = attn[b * 4 + 1];
    const float a2 = attn[b * 4 + 2];
    const float a3 = attn[b * 4 + 3];

    const size_t ks = (size_t)COUT * CIN * 9;             // stride per bank k
    const size_t wi = (((size_t)o * CIN) + c) * 9 + tap;  // weight[k][o][c][u][v]
    const float v = a0 * weight[wi]
                  + a1 * weight[wi + ks]
                  + a2 * weight[wi + 2 * ks]
                  + a3 * weight[wi + 3 * ks];
    aggw[idx] = f32_to_bf16(v);
}

// ---------------------------------------------------------------------------
// 4) implicit-GEMM conv via v_wmma_f32_16x16x32_bf16.
//    Block = 256 thr (8 wave32) computes 256(o) x 64(pixel) for one sample.
//    Flattened 72-step K loop (9 taps x 8 channel chunks of 32), software-
//    pipelined with scalarized registers and DOUBLE-BUFFERED LDS:
//    one barrier per step; global loads(it+1) overlap WMMAs(it); LDS stores
//    of step it+1 drain behind the matrix ops. All LDS traffic is 128-bit.
// ---------------------------------------------------------------------------
__global__ __launch_bounds__(256) void dynconv_wmma(const float* __restrict__ x,
                                                    const u16*  __restrict__ aggw,
                                                    const float* __restrict__ aggb,
                                                    float* __restrict__ out) {
    const int b  = blockIdx.y;
    const int n0 = blockIdx.x * 64;       // 49 tiles cover 3136 exactly

    __shared__ u16 As[2][256 * LDSTR];    // A tiles: [o][k] bf16 (2 x 20.5 KB)
    __shared__ u16 Bt[2][64 * LDSTR];     // B tiles transposed: [n][k] (2 x 5.1 KB)

    const int tid  = threadIdx.x;
    const int lane = tid & 31;
    const int wave = tid >> 5;
    const int mw     = (wave & 3) * 4;    // wave's first M subtile (of 16)
    const int n_sub0 = (wave >> 2) * 2;   // two 16-col N subtiles per wave

    F32x8 acc0_0, acc0_1, acc1_0, acc1_1, acc2_0, acc2_1, acc3_0, acc3_1;
#pragma unroll
    for (int r = 0; r < 8; ++r) {
        acc0_0.f[r] = 0.f; acc0_1.f[r] = 0.f;
        acc1_0.f[r] = 0.f; acc1_1.f[r] = 0.f;
        acc2_0.f[r] = 0.f; acc2_1.f[r] = 0.f;
        acc3_0.f[r] = 0.f; acc3_1.f[r] = 0.f;
    }

    // A staging map: thread -> rows (tid>>2)+64j, 8 consecutive k (uint4 ld/st)
    const int rowA = tid >> 2;            // 0..63 (+64j)
    const int kA   = (tid & 3) * 8;       // 0,8,16,24
    // B staging map: thread -> one pixel column nB, 8 consecutive k at kB0
    const int nB  = tid & 63;
    const int kB0 = (tid >> 6) * 8;       // 0,8,16,24
    const int py  = (n0 + nB) / HW;
    const int px  = (n0 + nB) % HW;

    const u16*   aggw_b = aggw + (size_t)b * COUT * 9 * CIN;
    const float* xb     = x + (size_t)b * CIN * NPIX;
    const size_t rstride = (size_t)9 * CIN;          // A row stride in u16

    // fragment geometry (CDNA5 16-bit WMMA lane striping):
    //   v16bf elements 0-7  = K kb..kb+7,      elements 8-15 = K 16+kb..16+kb+7
    //   kb = 0 for lanes 0-15, 8 for lanes 16-31  -> two ds_load_b128 per frag
    const int l15   = lane & 15;
    const int kb    = (lane < 16) ? 0 : 8;
    const int ncol0 = n_sub0 * 16 + l15;
    const int ncol1 = (n_sub0 + 1) * 16 + l15;

    // ---- pipeline registers: named scalars only (never spilled) ----
    uint4 a0, a1, a2, a3;
    float bb0, bb1, bb2, bb3, bb4, bb5, bb6, bb7;

    // ---- prologue: load step 0 (tap=0 -> u=0,v=0; chunk c0=0), fill buf 0 ----
    {
        const int iy = py - 1, ix = px - 1;
        const bool valid = (iy >= 0) && (iy < HW) && (ix >= 0) && (ix < HW);
        const float* xsrc = xb + (iy * HW + ix) + (size_t)kB0 * NPIX;
        const u16* ag = aggw_b + (size_t)rowA * rstride + kA;
        a0 = *(const uint4*)(ag);
        a1 = *(const uint4*)(ag +  64 * rstride);
        a2 = *(const uint4*)(ag + 128 * rstride);
        a3 = *(const uint4*)(ag + 192 * rstride);
        bb0 = valid ? xsrc[0 * NPIX] : 0.f;
        bb1 = valid ? xsrc[1 * NPIX] : 0.f;
        bb2 = valid ? xsrc[2 * NPIX] : 0.f;
        bb3 = valid ? xsrc[3 * NPIX] : 0.f;
        bb4 = valid ? xsrc[4 * NPIX] : 0.f;
        bb5 = valid ? xsrc[5 * NPIX] : 0.f;
        bb6 = valid ? xsrc[6 * NPIX] : 0.f;
        bb7 = valid ? xsrc[7 * NPIX] : 0.f;

        u16* asd = &As[0][rowA * LDSTR + kA];
        *(uint4*)(asd)               = a0;
        *(uint4*)(asd +  64 * LDSTR) = a1;
        *(uint4*)(asd + 128 * LDSTR) = a2;
        *(uint4*)(asd + 192 * LDSTR) = a3;
        uint4 bw;
        bw.x = pack_bf16(bb0, bb1);
        bw.y = pack_bf16(bb2, bb3);
        bw.z = pack_bf16(bb4, bb5);
        bw.w = pack_bf16(bb6, bb7);
        *(uint4*)&Bt[0][nB * LDSTR + kB0] = bw;
    }

#pragma unroll 1
    for (int it = 0; it < 72; ++it) {
        const u16* Ascur = &As[0][0] + (it & 1) * (256 * LDSTR);
        const u16* Btcur = &Bt[0][0] + (it & 1) * (64 * LDSTR);
        u16* Asnxt = &As[0][0] + ((it + 1) & 1) * (256 * LDSTR);
        u16* Btnxt = &Bt[0][0] + ((it + 1) & 1) * (64 * LDSTR);

        // single barrier per step: fences buf[cur] stores (it) and buf[nxt]
        // reads (it-1) across the workgroup
        __syncthreads();

        // ---- issue global loads for step it+1 (overlap with WMMAs below) ----
        const bool havenext = (it < 71);
        if (havenext) {
            const int nit  = it + 1;
            const int ntap = nit >> 3;
            const int c0   = (nit & 7) * 32;
            const int u = ntap / 3, v = ntap - 3 * u;
            const int iy = py + u - 1, ix = px + v - 1;
            const bool valid = (iy >= 0) && (iy < HW) && (ix >= 0) && (ix < HW);
            const float* xsrc = xb + (iy * HW + ix) + (size_t)(c0 + kB0) * NPIX;
            const u16* ag = aggw_b + (size_t)rowA * rstride
                          + (size_t)ntap * CIN + c0 + kA;
            a0 = *(const uint4*)(ag);
            a1 = *(const uint4*)(ag +  64 * rstride);
            a2 = *(const uint4*)(ag + 128 * rstride);
            a3 = *(const uint4*)(ag + 192 * rstride);
            bb0 = valid ? xsrc[0 * NPIX] : 0.f;
            bb1 = valid ? xsrc[1 * NPIX] : 0.f;
            bb2 = valid ? xsrc[2 * NPIX] : 0.f;
            bb3 = valid ? xsrc[3 * NPIX] : 0.f;
            bb4 = valid ? xsrc[4 * NPIX] : 0.f;
            bb5 = valid ? xsrc[5 * NPIX] : 0.f;
            bb6 = valid ? xsrc[6 * NPIX] : 0.f;
            bb7 = valid ? xsrc[7 * NPIX] : 0.f;
        }

        // ---- B fragments (reused across all 4 M subtiles), 2 x b128 each ----
        BF16x16 bf0, bf1;
        const u16* bp0 = Btcur + ncol0 * LDSTR;
        const u16* bp1 = Btcur + ncol1 * LDSTR;
        bf0.q[0] = *(const uint4*)(bp0 + kb);
        bf0.q[1] = *(const uint4*)(bp0 + 16 + kb);
        bf1.q[0] = *(const uint4*)(bp1 + kb);
        bf1.q[1] = *(const uint4*)(bp1 + 16 + kb);

        // ---- 4 A fragments, 8 matrix ops (explicit, no indexed arrays) ----
#define MTILE(J, ACC0, ACC1)                                                   \
        {                                                                      \
            BF16x16 af;                                                        \
            const u16* ap = Ascur + ((mw + (J)) * 16 + l15) * LDSTR;           \
            af.q[0] = *(const uint4*)(ap + kb);                                \
            af.q[1] = *(const uint4*)(ap + 16 + kb);                           \
            ACC0.v = __builtin_amdgcn_wmma_f32_16x16x32_bf16(                  \
                false, af.v, false, bf0.v, (short)0, ACC0.v, false, false);    \
            ACC1.v = __builtin_amdgcn_wmma_f32_16x16x32_bf16(                  \
                false, af.v, false, bf1.v, (short)0, ACC1.v, false, false);    \
        }
        MTILE(0, acc0_0, acc0_1)
        MTILE(1, acc1_0, acc1_1)
        MTILE(2, acc2_0, acc2_1)
        MTILE(3, acc3_0, acc3_1)
#undef MTILE

        // ---- commit staged registers (data of step it+1) to the other buffer
        if (havenext) {
            u16* asd = Asnxt + rowA * LDSTR + kA;
            *(uint4*)(asd)               = a0;
            *(uint4*)(asd +  64 * LDSTR) = a1;
            *(uint4*)(asd + 128 * LDSTR) = a2;
            *(uint4*)(asd + 192 * LDSTR) = a3;
            uint4 bw;
            bw.x = pack_bf16(bb0, bb1);
            bw.y = pack_bf16(bb2, bb3);
            bw.z = pack_bf16(bb4, bb5);
            bw.w = pack_bf16(bb6, bb7);
            *(uint4*)(Btnxt + nB * LDSTR + kB0) = bw;
        }
    }

    // ---- epilogue: D layout -> global (M = r + (lane<16?0:8), N = lane&15) ----
    const int rbase = (lane < 16) ? 0 : 8;
    const int nn0 = n0 + n_sub0 * 16 + l15;
    const int nn1 = nn0 + 16;
    const int oy0 = nn0 / HW, ox0 = nn0 % HW;
    const int oy1 = nn1 / HW, ox1 = nn1 % HW;

#define EPI(J, ACC0, ACC1)                                                     \
    {                                                                          \
        const int obase = (mw + (J)) * 16 + rbase;                             \
        _Pragma("unroll")                                                      \
        for (int r = 0; r < 8; ++r) {                                          \
            const int o = obase + r;                                           \
            const float bv = aggb[b * COUT + o];                               \
            float* orow = out + (size_t)(b * COUT + o) * NPIX;                 \
            orow[oy0 * HW + ox0] = ACC0.f[r] + bv;                             \
            orow[oy1 * HW + ox1] = ACC1.f[r] + bv;                             \
        }                                                                      \
    }
    EPI(0, acc0_0, acc0_1)
    EPI(1, acc1_0, acc1_1)
    EPI(2, acc2_0, acc2_1)
    EPI(3, acc3_0, acc3_1)
#undef EPI
}

// ---------------------------------------------------------------------------
extern "C" void kernel_launch(void* const* d_in, const int* in_sizes, int n_in,
                              void* d_out, int out_size, void* d_ws, size_t ws_size,
                              hipStream_t stream) {
    (void)in_sizes; (void)n_in; (void)out_size; (void)ws_size;
    const float* x      = (const float*)d_in[0];
    const float* weight = (const float*)d_in[1];
    const float* bias_k = (const float*)d_in[2];
    const float* w1     = (const float*)d_in[3];
    const float* w2     = (const float*)d_in[4];
    const float* b2     = (const float*)d_in[5];
    float* out = (float*)d_out;

    char* ws = (char*)d_ws;
    float* p    = (float*)(ws);             // 32*512 f32   = 65536 B
    float* attn = (float*)(ws + 65536);     // 32*4   f32   = 512 B
    float* aggb = (float*)(ws + 66048);     // 32*256 f32   = 32768 B
    u16*   aggw = (u16*)  (ws + 98816);     // 32*256*9*256 bf16 ~= 36 MB

    pool_kernel<<<B_SZ * CIN, 256, 0, stream>>>(x, p);
    attn_kernel<<<1, 256, 0, stream>>>(p, w1, w2, b2, bias_k, attn, aggb);

    const long long naggw = (long long)B_SZ * COUT * 9 * CIN;   // 18,874,368
    aggw_kernel<<<(unsigned)(naggw / 256), 256, 0, stream>>>(weight, attn, aggw);

    dim3 grid(NPIX / 64, B_SZ);             // (49, 32)
    dynconv_wmma<<<grid, 256, 0, stream>>>(x, aggw, aggb, out);
}
